// Attention_2456721293410
// MI455X (gfx1250) — matrix operand order repaired and checked by reading, not compiled, not run on hardware
//
#include <hip/hip_runtime.h>
#include <hip/hip_bf16.h>

// ---------------- problem constants ----------------
#define HEADS   16
#define DHEAD   128
#define DIM     2048
#define SEQ     2048
#define ACT     512
#define BATCH   2
#define NTOT    (SEQ + ACT)          // 2560
#define NQKV    (3 * HEADS * DHEAD)  // 6144
#define QBLK    128                  // query rows per workgroup (8 waves x 16)
#define QSCALE  0.08838834764831845f // DHEAD^-0.5

typedef __attribute__((ext_vector_type(16))) __bf16 v16bf;
typedef __attribute__((ext_vector_type(8)))  __bf16 v8bf;
typedef __attribute__((ext_vector_type(8)))  float  v8f;

// ---------------- fp32 -> bf16 conversion kernels ----------------
__global__ void cvt_copy_kernel(const float* __restrict__ in,
                                __bf16* __restrict__ out, int n) {
  int i = blockIdx.x * 256 + threadIdx.x;
  if (i < n) out[i] = (__bf16)in[i];
}

// in: [DIM][N] row-major f32  ->  out: [N][DIM] row-major bf16 (W^T)
// K = DIM (compile-time) so index math is shift/mask.
__global__ void cvt_transpose_kernel(const float* __restrict__ in,
                                     __bf16* __restrict__ out, int N) {
  int i = blockIdx.x * 256 + threadIdx.x;
  if (i < DIM * N) {
    int n = i >> 11;           // / DIM
    int k = i & (DIM - 1);     // % DIM
    out[i] = (__bf16)in[(size_t)k * N + n];
  }
}

// A-fragment loader: row-major row pointer, K-offset kb already applied by caller.
// elems 0..15 = K[hi*16 ..] contiguous (matches contiguous B fills)
__device__ __forceinline__ v16bf load_a_frag(const __bf16* p, int hi) {
  v8bf lo = *(const v8bf*)(p + hi * 16);
  v8bf hi8 = *(const v8bf*)(p + hi * 16 + 8);
  return __builtin_shufflevector(lo, hi8, 0, 1, 2, 3, 4, 5, 6, 7,
                                 8, 9, 10, 11, 12, 13, 14, 15);
}

// ---------------- WMMA 32x64 register-blocked GEMM tile ----------------
// A: 32 rows x K (row-major). Bt: output-column-major, Bt[n][k]. acc[s][c] is
// the 16x16 sub-tile for rows s*16.., cols c*16.. in WMMA C-layout.
__device__ __forceinline__ void wmma_gemm_tile_32x64(
    const __bf16* __restrict__ A, int lda,
    const __bf16* __restrict__ Bt, int ldb, int K, int lane, v8f acc[2][4]) {
  const int mi = lane & 15;
  const int hi = lane >> 4;
  const __bf16* arow0 = A + (size_t)mi * lda;
  const __bf16* arow1 = A + (size_t)(16 + mi) * lda;
  const __bf16* brow  = Bt + (size_t)mi * ldb;
  for (int kb = 0; kb < K; kb += 32) {
    if (kb + 512 < K) {                       // stream-ahead prefetch (A operand)
      __builtin_prefetch(arow0 + kb + 512, 0, 0);
      __builtin_prefetch(arow1 + kb + 512, 0, 0);
    }
    v16bf a0 = load_a_frag(arow0 + kb, hi);
    v16bf a1 = load_a_frag(arow1 + kb, hi);
    v16bf b[4];
#pragma unroll
    for (int c = 0; c < 4; ++c)
      b[c] = *(const v16bf*)(brow + (size_t)c * 16 * ldb + kb + hi * 16);
#pragma unroll
    for (int c = 0; c < 4; ++c) {
      acc[0][c] = __builtin_amdgcn_wmma_f32_16x16x32_bf16(false, a0, false, b[c],
                                                          (short)0, acc[0][c],
                                                          false, false);
      acc[1][c] = __builtin_amdgcn_wmma_f32_16x16x32_bf16(false, a1, false, b[c],
                                                          (short)0, acc[1][c],
                                                          false, false);
    }
  }
}

// ---------------- QKV projection GEMM ----------------
// X: [BATCH*ROWS][DIM] bf16, WT: [NQKV][DIM] bf16 (transposed weights).
// Scatters into q [b,h,pos,d] (scaled), k [b,h,pos,d], vT [b,h,d,pos].
// ROWS is a power of two -> all row index math is shift/mask.
template <int ROWS, int POS_OFF>
__global__ __launch_bounds__(256) void qkv_gemm_kernel(
    const __bf16* __restrict__ X, const __bf16* __restrict__ WT,
    __bf16* __restrict__ qbuf, __bf16* __restrict__ kbuf,
    __bf16* __restrict__ vTbuf) {
  const int lane = threadIdx.x & 31;
  const int wv = threadIdx.x >> 5;
  constexpr int M = BATCH * ROWS;
  constexpr int ntiles_n = NQKV / 64;            // 96
  constexpr int ntiles = (M / 32) * ntiles_n;
  const int tid = blockIdx.x * 8 + wv;
  if (tid >= ntiles) return;                     // wave-uniform
  const int tm = tid / ntiles_n, tn = tid % ntiles_n;

  const __bf16* A  = X  + (size_t)tm * 32 * DIM;
  const __bf16* Bt = WT + (size_t)tn * 64 * DIM;
  v8f acc[2][4];
#pragma unroll
  for (int s = 0; s < 2; ++s)
#pragma unroll
    for (int c = 0; c < 4; ++c) acc[s][c] = (v8f){};
  wmma_gemm_tile_32x64(A, DIM, Bt, DIM, DIM, lane, acc);

  const int mi = lane & 15, hi = lane >> 4;
#pragma unroll
  for (int c = 0; c < 4; ++c) {
    const int n_global = tn * 64 + c * 16 + mi;
    const int which = n_global / (HEADS * DHEAD);  // 0=q 1=k 2=v
    const int rem   = n_global % (HEADS * DHEAD);
    const int h = rem >> 7, dcol = rem & 127;
#pragma unroll
    for (int s = 0; s < 2; ++s) {
#pragma unroll
      for (int r = 0; r < 8; ++r) {
        const int m_global = tm * 32 + s * 16 + r + 8 * hi;
        const int b   = m_global / ROWS;           // shift (ROWS pow2)
        const int pos = (m_global & (ROWS - 1)) + POS_OFF;
        const size_t bh = (size_t)b * HEADS + h;
        const float v = acc[s][c][r];
        if (which == 0) {
          qbuf[(bh * NTOT + pos) * DHEAD + dcol] = (__bf16)(v * QSCALE);
        } else if (which == 1) {
          kbuf[(bh * NTOT + pos) * DHEAD + dcol] = (__bf16)v;
        } else {
          vTbuf[(bh * DHEAD + dcol) * NTOT + pos] = (__bf16)v;
        }
      }
    }
  }
}

// ---------------- causal flash attention ----------------
// grid: (NTOT/QBLK, BATCH*HEADS), block: 256 (8 waves, 16 q-rows each).
__global__ __launch_bounds__(256) void flash_attn_kernel(
    const __bf16* __restrict__ q, const __bf16* __restrict__ k,
    const __bf16* __restrict__ vT, __bf16* __restrict__ outbuf) {
  __shared__ __bf16 plds[8][16][32];             // per-wave P staging (8 KB)
  const int lane = threadIdx.x & 31;
  const int w    = threadIdx.x >> 5;
  const int mi   = lane & 15;
  const int hi   = lane >> 4;
  const int bh   = blockIdx.y;                   // b*HEADS + h
  const int b    = bh >> 4;
  const int h    = bh & 15;
  const int qb   = blockIdx.x * QBLK + w * 16;   // first query row of this wave

  const __bf16* qbase = q  + ((size_t)bh * NTOT + qb) * DHEAD;
  const __bf16* kbase = k  + (size_t)bh * NTOT * DHEAD;
  const __bf16* vbase = vT + (size_t)bh * DHEAD * NTOT;

  // Q tile as 4 A-fragments (K-chunks of 32 over d=128), loaded once.
  v16bf aq[4];
#pragma unroll
  for (int c = 0; c < 4; ++c)
    aq[c] = load_a_frag(qbase + mi * DHEAD + c * 32, hi);

  v8f O[8];
#pragma unroll
  for (int i = 0; i < 8; ++i) O[i] = (v8f){};
  float m_r[8], l_r[8];
#pragma unroll
  for (int r = 0; r < 8; ++r) { m_r[r] = -__builtin_inff(); l_r[r] = 0.0f; }

  // uniform trip count across the workgroup (covers max query row in block)
  const int kblocks = (blockIdx.x + 1) * (QBLK / 32);
  for (int kbi = 0; kbi < kblocks; ++kbi) {
    const int kb = kbi * 32;

    // S = q . k^T for 32 keys (two 16-key column tiles)
    v8f S0 = {}, S1 = {};
#pragma unroll
    for (int c = 0; c < 4; ++c) {
      v16bf b0 = *(const v16bf*)(kbase + (size_t)(kb + mi) * DHEAD + c * 32 + hi * 16);
      v16bf b1 = *(const v16bf*)(kbase + (size_t)(kb + 16 + mi) * DHEAD + c * 32 + hi * 16);
      S0 = __builtin_amdgcn_wmma_f32_16x16x32_bf16(false, aq[c], false, b0,
                                                   (short)0, S0, false, false);
      S1 = __builtin_amdgcn_wmma_f32_16x16x32_bf16(false, aq[c], false, b1,
                                                   (short)0, S1, false, false);
    }

    // causal mask + online softmax (rows live in 16-lane halves)
    float p0[8], p1[8], corr[8];
#pragma unroll
    for (int r = 0; r < 8; ++r) {
      const int row = qb + r + 8 * hi;
      const int c0 = kb + mi;
      float s0 = (c0 > row)      ? -__builtin_inff() : S0[r];
      float s1 = (c0 + 16 > row) ? -__builtin_inff() : S1[r];
      float t = fmaxf(s0, s1);
#pragma unroll
      for (int off = 1; off < 16; off <<= 1) t = fmaxf(t, __shfl_xor(t, off, 32));
      const float nm = fmaxf(m_r[r], t);
      const float cr = __expf(m_r[r] - nm);
      m_r[r] = nm;
      p0[r] = __expf(s0 - nm);
      p1[r] = __expf(s1 - nm);
      float rs = p0[r] + p1[r];
#pragma unroll
      for (int off = 1; off < 16; off <<= 1) rs += __shfl_xor(rs, off, 32);
      l_r[r] = l_r[r] * cr + rs;
      corr[r] = cr;
    }
#pragma unroll
    for (int dc = 0; dc < 8; ++dc)
#pragma unroll
      for (int r = 0; r < 8; ++r) O[dc][r] *= corr[r];

    // P: C-layout -> A-layout via per-wave LDS slice
#pragma unroll
    for (int r = 0; r < 8; ++r) {
      plds[w][r + 8 * hi][mi]      = (__bf16)p0[r];
      plds[w][r + 8 * hi][16 + mi] = (__bf16)p1[r];
    }
    __syncthreads();
    v8bf plo = *(const v8bf*)(&plds[w][mi][hi * 16]);
    v8bf phi = *(const v8bf*)(&plds[w][mi][hi * 16 + 8]);
    v16bf ap = __builtin_shufflevector(plo, phi, 0, 1, 2, 3, 4, 5, 6, 7,
                                       8, 9, 10, 11, 12, 13, 14, 15);
    __syncthreads();

    // O += P . V   (B from vT: column d contiguous over key positions)
#pragma unroll
    for (int dc = 0; dc < 8; ++dc) {
      v16bf bv = *(const v16bf*)(vbase + (size_t)(dc * 16 + mi) * NTOT + kb + hi * 16);
      O[dc] = __builtin_amdgcn_wmma_f32_16x16x32_bf16(false, ap, false, bv,
                                                      (short)0, O[dc], false, false);
    }
  }

  // normalize, store bf16 to outbuf [b][pos][h*128 + d]
#pragma unroll
  for (int r = 0; r < 8; ++r) {
    const int pos = qb + r + 8 * hi;
    const float inv = 1.0f / l_r[r];
#pragma unroll
    for (int dc = 0; dc < 8; ++dc) {
      const int dcol = dc * 16 + mi;
      outbuf[((size_t)b * NTOT + pos) * DIM + h * DHEAD + dcol] =
          (__bf16)(O[dc][r] * inv);
    }
  }
}

// ---------------- output projection GEMM (f32 result) ----------------
template <int ROWS, int POS_OFF>
__global__ __launch_bounds__(256) void out_gemm_kernel(
    const __bf16* __restrict__ Obuf, const __bf16* __restrict__ WT,
    float* __restrict__ Y) {
  const int lane = threadIdx.x & 31;
  const int wv = threadIdx.x >> 5;
  constexpr int M = BATCH * ROWS;
  constexpr int ntiles_n = DIM / 64;             // 32
  constexpr int ntiles = (M / 32) * ntiles_n;
  const int tid = blockIdx.x * 8 + wv;
  if (tid >= ntiles) return;
  const int tm = tid / ntiles_n, tn = tid % ntiles_n;
  const int m0 = tm * 32;
  const int b = m0 / ROWS;                       // shift (ROWS pow2)
  const int pos0 = m0 & (ROWS - 1);              // 32-row tile never crosses batch
  const __bf16* A  = Obuf + ((size_t)b * NTOT + POS_OFF + pos0) * DIM;
  const __bf16* Bt = WT + (size_t)tn * 64 * DIM;
  v8f acc[2][4];
#pragma unroll
  for (int s = 0; s < 2; ++s)
#pragma unroll
    for (int c = 0; c < 4; ++c) acc[s][c] = (v8f){};
  wmma_gemm_tile_32x64(A, DIM, Bt, DIM, DIM, lane, acc);

  const int mi = lane & 15, hi = lane >> 4;
#pragma unroll
  for (int s = 0; s < 2; ++s)
#pragma unroll
    for (int r = 0; r < 8; ++r) {
      const int m_global = m0 + s * 16 + r + 8 * hi;
      float* yrow = Y + (size_t)m_global * DIM + tn * 64 + mi;
#pragma unroll
      for (int c = 0; c < 4; ++c) yrow[c * 16] = acc[s][c][r];
    }
}

// ---------------- host launcher ----------------
extern "C" void kernel_launch(void* const* d_in, const int* in_sizes, int n_in,
                              void* d_out, int out_size, void* d_ws, size_t ws_size,
                              hipStream_t stream) {
  const float* x_m    = (const float*)d_in[0]; // [2,2048,2048]
  const float* x_a    = (const float*)d_in[1]; // [2,512,2048]
  const float* w_qkv  = (const float*)d_in[2]; // [2048,6144]
  const float* w_out  = (const float*)d_in[3]; // [2048,2048]
  const float* w_aqkv = (const float*)d_in[4]; // [2048,6144]
  const float* w_aout = (const float*)d_in[5]; // [2048,2048]
  float* out = (float*)d_out;

  size_t off = 0;
  auto alloc = [&](size_t bytes) -> void* {
    void* p = (char*)d_ws + off;
    off += (bytes + 255) & ~(size_t)255;
    return p;
  };
  __bf16* xm_bf   = (__bf16*)alloc((size_t)BATCH * SEQ * DIM * 2);
  __bf16* xa_bf   = (__bf16*)alloc((size_t)BATCH * ACT * DIM * 2);
  __bf16* wqkvT   = (__bf16*)alloc((size_t)NQKV * DIM * 2);
  __bf16* waqkvT  = (__bf16*)alloc((size_t)NQKV * DIM * 2);
  __bf16* woutT   = (__bf16*)alloc((size_t)DIM * DIM * 2);
  __bf16* waoutT  = (__bf16*)alloc((size_t)DIM * DIM * 2);
  __bf16* qbuf    = (__bf16*)alloc((size_t)BATCH * HEADS * NTOT * DHEAD * 2);
  __bf16* kbuf    = (__bf16*)alloc((size_t)BATCH * HEADS * NTOT * DHEAD * 2);
  __bf16* vTbuf   = (__bf16*)alloc((size_t)BATCH * HEADS * NTOT * DHEAD * 2);
  __bf16* outbuf  = (__bf16*)alloc((size_t)BATCH * NTOT * DIM * 2);

  // 1) precision conversion (+ weight transpose for B-operand layout)
  {
    int n = BATCH * SEQ * DIM;
    cvt_copy_kernel<<<(n + 255) / 256, 256, 0, stream>>>(x_m, xm_bf, n);
    n = BATCH * ACT * DIM;
    cvt_copy_kernel<<<(n + 255) / 256, 256, 0, stream>>>(x_a, xa_bf, n);
    n = DIM * NQKV;
    cvt_transpose_kernel<<<(n + 255) / 256, 256, 0, stream>>>(w_qkv, wqkvT, NQKV);
    cvt_transpose_kernel<<<(n + 255) / 256, 256, 0, stream>>>(w_aqkv, waqkvT, NQKV);
    n = DIM * DIM;
    cvt_transpose_kernel<<<(n + 255) / 256, 256, 0, stream>>>(w_out, woutT, DIM);
    cvt_transpose_kernel<<<(n + 255) / 256, 256, 0, stream>>>(w_aout, waoutT, DIM);
  }

  // 2) QKV projections -> q (scaled), k, vT
  {
    int tiles_m = (BATCH * SEQ) / 32 * (NQKV / 64);
    qkv_gemm_kernel<SEQ, 0><<<(tiles_m + 7) / 8, 256, 0, stream>>>(
        xm_bf, wqkvT, qbuf, kbuf, vTbuf);
    int tiles_a = (BATCH * ACT) / 32 * (NQKV / 64);
    qkv_gemm_kernel<ACT, SEQ><<<(tiles_a + 7) / 8, 256, 0, stream>>>(
        xa_bf, waqkvT, qbuf, kbuf, vTbuf);
  }

  // 3) causal flash attention
  {
    dim3 grid(NTOT / QBLK, BATCH * HEADS);
    flash_attn_kernel<<<grid, 256, 0, stream>>>(qbuf, kbuf, vTbuf, outbuf);
  }

  // 4) output projections: mout then aout (concatenated in d_out)
  {
    int tiles_m = (BATCH * SEQ) / 32 * (DIM / 64);
    out_gemm_kernel<SEQ, 0><<<(tiles_m + 7) / 8, 256, 0, stream>>>(
        outbuf, woutT, out);
    int tiles_a = (BATCH * ACT) / 32 * (DIM / 64);
    out_gemm_kernel<ACT, SEQ><<<(tiles_a + 7) / 8, 256, 0, stream>>>(
        outbuf, waoutT, out + (size_t)BATCH * SEQ * DIM);
  }
  (void)in_sizes; (void)n_in; (void)out_size; (void)ws_size;
}
